// ConvGeodesic_74749610819723
// MI455X (gfx1250) — compile-verified, matrix operand order checked by reading, not run on hardware
//
#include <hip/hip_runtime.h>
#include <hip/hip_bf16.h>

// ---- problem constants (from reference) ----
#define BB 4
#define NN 10000
#define KR 5
#define KA 8
#define CC 32
#define OO 64
#define KK 2

#define KDIM (KR * KA * CC)   // 1280  (GEMM K)
#define COLS (KA * OO)        // 512   (GEMM N = rotations x out-channels)
#define MROWS (BB * NN)       // 40000 (GEMM M)
#define MTILE 64              // rows per workgroup (40000 / 64 = 625 exact)
#define NKT (KDIM / 32)       // 40 k-tiles of 32
#define NCT (COLS / 16)       // 32 column tiles of 16
#define BCHUNK_KT 2           // k-tiles of B staged in LDS per outer iter

#define APAD 8                        // +16B per LDS A row -> conflict-free ds_load_b128
#define ASTRIDE (KDIM + APAD)         // 1288 halves
#define LDS_A_HALVES (MTILE * ASTRIDE)        // 82432 halves = 164,864 B
#define LDS_B_HALVES (BCHUNK_KT * 32 * 512)   // 32768 halves =  65,536 B  (total 230,400 B < 320 KB)

typedef __attribute__((ext_vector_type(16))) _Float16 v16h;
typedef __attribute__((ext_vector_type(8)))  _Float16 v8h;
typedef __attribute__((ext_vector_type(4)))  _Float16 v4h;
typedef __attribute__((ext_vector_type(8)))  float    v8f;
typedef __attribute__((ext_vector_type(4)))  float    v4f;

// ------------------------------------------------------------------
// Prep: expand W[(p,q,c),(r,o)] = Ksum[p,(q+r)%8,o,c] into f16 blobs laid
// out exactly as WMMA B fragments (32x16 tile = 32 lanes x 32 bytes = 1 KB),
// with per-lane 16B-chunk XOR swizzle for bank-conflict-free LDS reads.
// ------------------------------------------------------------------
__global__ __launch_bounds__(256) void ConvGeo_prep_w(const float* __restrict__ kern,
                                                      _Float16* __restrict__ wt) {
  int t = blockIdx.x * 256 + threadIdx.x;
  if (t >= KDIM * COLS) return;
  int h    = t & 15;          // half index within lane's 32B
  int L    = (t >> 4) & 31;   // lane
  int blob = t >> 9;          // fragment blob (512 halves)
  int kt = blob >> 5;         // k-tile 0..39
  int j  = blob & 31;         // column tile 0..31
  int col = j * 16 + (L & 15);
  int r = col >> 6;           // rotation
  int o = col & 63;           // out channel
  int sw = (L >> 3) & 1;      // bank swizzle: swap the two 16B chunks
  int s  = (h >> 3) ^ sw;     // logical chunk (0: k+0..7, 1: k+16..23)
  int kk = (h & 7) + (s ? 16 : 0) + ((L < 16) ? 0 : 8);
  int k  = kt * 32 + kk;      // global K index = (p*8+q)*32 + c
  int pq = k >> 5;
  int c  = k & 31;
  int p  = pq >> 3;
  int q  = pq & 7;
  int q2 = (q + r) & 7;       // rotated angular bin
  long base = (((long)p * KA + q2) * OO + o) * CC + c;
  float v = kern[base] + kern[(long)KR * KA * OO * CC + base];  // sum over K=2 kernels
  wt[t] = (_Float16)v;
}

__global__ void ConvGeo_prep_bias(const float* __restrict__ bias, float* __restrict__ bsum) {
  int o = threadIdx.x;  // 64 threads
  float s = 0.f;
  for (int pq = 0; pq < KR * KA; ++pq) s += bias[pq * OO + o];
  bsum[o] = (float)KK * s;
}

// ------------------------------------------------------------------
// Main fused kernel: gather+interp -> LDS (f16), then WMMA GEMM.
// 256 threads = 8 waves; wave w: row-tile = w&3, column half = w>>2.
// ------------------------------------------------------------------
__global__ __launch_bounds__(256) void ConvGeo_main(const float* __restrict__ signal,
                                                    const float* __restrict__ bary,
                                                    const _Float16* __restrict__ wt,
                                                    const float* __restrict__ bsum,
                                                    float* __restrict__ out) {
  __shared__ _Float16 lds[LDS_A_HALVES + LDS_B_HALVES];
  _Float16* ldsA = lds;
  _Float16* ldsB = lds + LDS_A_HALVES;

  const int tid   = threadIdx.x;
  const int wgRow = blockIdx.x * MTILE;

  // -------- Phase 1: barycentric gather + interpolation into LDS A (f16) ----
  // 2560 (row, p*q) tasks / 256 threads = 10 each; zero divergence.
  for (int task = tid; task < MTILE * KR * KA; task += 256) {
    int rl = task / (KR * KA);
    int pq = task - rl * (KR * KA);
    int row = wgRow + rl;
    int b = row / NN;
    int n = row - b * NN;
    const float* bc = bary + ((long)row * (KR * KA) + pq) * 6;
    int   i0 = (int)bc[0]; float w0 = bc[1];
    int   i1 = (int)bc[2]; float w1 = bc[3];
    int   i2 = (int)bc[4]; float w2 = bc[5];
    const float* s0 = signal + ((long)b * NN + i0) * CC;
    const float* s1 = signal + ((long)b * NN + i1) * CC;
    const float* s2 = signal + ((long)b * NN + i2) * CC;
    _Float16* dst = ldsA + rl * ASTRIDE + pq * CC;
#pragma unroll
    for (int c = 0; c < CC; c += 4) {
      v4f a0 = *(const v4f*)(s0 + c);
      v4f a1 = *(const v4f*)(s1 + c);
      v4f a2 = *(const v4f*)(s2 + c);
      v4f rr = a0 * w0 + a1 * w1 + a2 * w2;
      v4h hv = {(_Float16)rr[0], (_Float16)rr[1], (_Float16)rr[2], (_Float16)rr[3]};
      *(v4h*)(dst + c) = hv;  // ds_store_b64
    }
    (void)n;
  }
  __syncthreads();

  // -------- Phase 2: WMMA GEMM --------
  const int wave = tid >> 5;
  const int lane = tid & 31;
  const int rt   = wave & 3;    // row tile (16 rows) 0..3
  const int half = wave >> 2;   // column half: 16 col-tiles each
  const int mloc  = rt * 16 + (lane & 15);
  const int kbase = (lane < 16) ? 0 : 8;
  const int sw    = (lane >> 3) & 1;  // B fragment bank swizzle

  v8f acc[16];
  const v8f zero = {0.f, 0.f, 0.f, 0.f, 0.f, 0.f, 0.f, 0.f};
#pragma unroll
  for (int j = 0; j < 16; ++j) acc[j] = zero;

  for (int ko = 0; ko < NKT; ko += BCHUNK_KT) {
    __syncthreads();  // previous chunk's reads done before overwrite
    {   // cooperative copy of 64 KB of B blobs (L2-resident) into LDS
      const v8h* src  = (const v8h*)(wt + (long)ko * 32 * 512);
      v8h*       dstB = (v8h*)ldsB;
#pragma unroll
      for (int i = 0; i < 16; ++i) dstB[tid + i * 256] = src[tid + i * 256];
    }
    __syncthreads();

#pragma unroll
    for (int kt = 0; kt < BCHUNK_KT; ++kt) {
      // A fragment: 16x32 f16, lane<16 holds K {0..7,16..23}, lane>=16 {8..15,24..31}
      const _Float16* ap = ldsA + mloc * ASTRIDE + (ko + kt) * 32 + kbase;
      v8h alo = *(const v8h*)ap;         // ds_load_b128
      v8h ahi = *(const v8h*)(ap + 16);  // ds_load_b128
      v16h afrag = __builtin_shufflevector(alo, ahi,
          0, 1, 2, 3, 4, 5, 6, 7, 8, 9, 10, 11, 12, 13, 14, 15);
#pragma unroll
      for (int j = 0; j < 16; ++j) {
        const _Float16* bp = ldsB + ((kt * 32 + half * 16 + j) * 512) + lane * 16;
        v8h blo = *(const v8h*)(bp + sw * 8);        // logical K {0..7}+base
        v8h bhi = *(const v8h*)(bp + (1 - sw) * 8);  // logical K {16..23}+base
        v16h bfrag = __builtin_shufflevector(blo, bhi,
            0, 1, 2, 3, 4, 5, 6, 7, 8, 9, 10, 11, 12, 13, 14, 15);
        acc[j] = __builtin_amdgcn_wmma_f32_16x16x32_f16(
            false, afrag, false, bfrag, (short)0, acc[j], false, false);
      }
    }
  }

  // -------- Epilogue: bias + scatter to out[B, R, N, O] --------
#pragma unroll
  for (int j = 0; j < 16; ++j) {
    int col = (half * 16 + j) * 16 + (lane & 15);
    int r = col >> 6;
    int o = col & 63;
    float bv = bsum[o];
    int mbase = wgRow + rt * 16 + ((lane < 16) ? 0 : 8);
#pragma unroll
    for (int v = 0; v < 8; ++v) {
      int row = mbase + v;  // D layout: VGPR v -> M = v (+8 for upper lanes)
      int b = row / NN;
      int n = row - b * NN;
      out[(((long)b * KA + r) * NN + n) * OO + o] = acc[j][v] + bv;
    }
  }
}

extern "C" void kernel_launch(void* const* d_in, const int* in_sizes, int n_in,
                              void* d_out, int out_size, void* d_ws, size_t ws_size,
                              hipStream_t stream) {
  (void)in_sizes; (void)n_in; (void)out_size; (void)ws_size;
  const float* signal = (const float*)d_in[0];
  const float* bary   = (const float*)d_in[1];
  const float* kern   = (const float*)d_in[2];
  const float* bias   = (const float*)d_in[3];
  float*       out    = (float*)d_out;

  _Float16* wt   = (_Float16*)d_ws;                                   // 1280*512*2 = 1.31 MB
  float*    bsum = (float*)((char*)d_ws + (size_t)KDIM * COLS * 2);   // 64 floats

  ConvGeo_prep_w<<<(KDIM * COLS + 255) / 256, 256, 0, stream>>>(kern, wt);
  ConvGeo_prep_bias<<<1, 64, 0, stream>>>(bias, bsum);
  ConvGeo_main<<<MROWS / MTILE, 256, 0, stream>>>(signal, bary, wt, bsum, out);
}
